// HelixNeuralNetwork_15925738734064
// MI455X (gfx1250) — matrix operand order repaired and checked by reading, not compiled, not run on hardware
//
#include <hip/hip_runtime.h>
#include <math.h>

typedef __attribute__((ext_vector_type(16))) _Float16 v16h;
typedef __attribute__((ext_vector_type(8)))  _Float16 h8;
typedef __attribute__((ext_vector_type(8)))  float    v8f;

#define IN_DIM   512
#define OUT_DIM  128
#define NOSC     256
#define NLAYERS  4
#define NSTEPS   10
#define DTC      0.1f
#define BM       16         // batch rows per block (1 M-tile)
#define NTHREADS 256        // 8 waves -> 2 waves/SIMD for co-execution
#define LDSTRIDE 264        // f16 row stride for S/C tiles (pad vs 256 to spread banks)

// d_ws layout in _Float16 units (total 425984 halves = 832 KB):
//   encB : [16 kt][16 nt][32 lane][16]  W_enc^T packed, 131072 halves @ 0
//   KB   : 4 layers x [8 kt][16 nt][32][16] K_eff packed, 262144 halves @ 131072
//   decB : [8 kt][8 nt][32][16]         W_dec^T packed, 32768  halves @ 393216
#define ENCB_OFF 0
#define KB_OFF   131072
#define KB_LAYER 65536
#define DECB_OFF 393216
#define WS_HALVES 425984

__device__ __forceinline__ float wrapf(float t) {
  // maps to (-pi, pi] (reference uses atan2(sin,cos); equivalent up to boundary)
  return t - 6.28318530717958647692f * rintf(t * 0.15915494309189533577f);
}

__device__ __forceinline__ v8f wmma_f16(v16h a, v16h b, v8f c) {
  return __builtin_amdgcn_wmma_f32_16x16x32_f16(false, a, false, b, (short)0, c, false, false);
}

// A-tile loader: 16x32 f16 A-layout. lane m(0..15) row M=m, khalf = lane>>4 selects
// K = khalf*16 + 0..15 (elements 0..15).
__device__ __forceinline__ v16h load_A(const _Float16* base, int mrow, int kb, int khalf) {
  const _Float16* p = base + mrow * LDSTRIDE + kb + khalf * 16;
  union { v16h v; h8 h[2]; } u;
  u.h[0] = *(const h8*)(p);
  u.h[1] = *(const h8*)(p + 8);
  return u.v;
}

// B-tile loader from packed buffer (global or LDS): 32 B contiguous per lane
__device__ __forceinline__ v16h load_B(const _Float16* g, int tile, int lane) {
  union { v16h v; h8 h[2]; } u;
  const _Float16* p = g + ((size_t)tile * 32 + lane) * 16;
  u.h[0] = *(const h8*)(p);
  u.h[1] = *(const h8*)(p + 8);
  return u.v;
}

// ---------------- prep: pack B operands into WMMA register layout (f16) -------------
__global__ __launch_bounds__(256) void helix_prep(const float* __restrict__ W_enc,
                                                  const float* __restrict__ Ks,
                                                  const float* __restrict__ mu_gates,
                                                  const float* __restrict__ W_dec,
                                                  _Float16* __restrict__ ws) {
  int p = blockIdx.x * 256 + threadIdx.x;
  if (p >= WS_HALVES) return;
  if (p < KB_OFF) {
    // encB: B[k][n] = W_enc[n][k], KT=16, NT=16
    int e = p & 15, lane = (p >> 4) & 31, nt = (p >> 9) & 15, kt = p >> 13;
    int n = nt * 16 + (lane & 15);
    int k = kt * 32 + (lane >> 4) * 16 + e;
    ws[p] = (_Float16)W_enc[n * IN_DIM + k];
  } else if (p < DECB_OFF) {
    // KB: B[k][n] = K_eff[k][n] (K symmetric -> K^T == K); fold mu_gate*0.5
    int q = p - KB_OFF;
    int L = q >> 16, pp = q & 65535;
    int e = pp & 15, lane = (pp >> 4) & 31, nt = (pp >> 9) & 15, kt = pp >> 13;
    int n = nt * 16 + (lane & 15);
    int k = kt * 32 + (lane >> 4) * 16 + e;
    float scale = mu_gates[L] * 0.5f;   // MU_GATE_FACTOR
    ws[p] = (_Float16)(Ks[(size_t)L * NOSC * NOSC + k * NOSC + n] * scale);
  } else {
    // decB: B[k][o] = W_dec[o][k], KT=8, NT=8
    int q = p - DECB_OFF;
    int e = q & 15, lane = (q >> 4) & 31, nt = (q >> 9) & 7, kt = q >> 12;
    int o = nt * 16 + (lane & 15);
    int k = kt * 32 + (lane >> 4) * 16 + e;
    ws[p] = (_Float16)W_dec[o * NOSC + k];
  }
}

// ---------------- main: encoder + 4x10 Kuramoto steps + coherence + decoder ---------
__global__ __launch_bounds__(NTHREADS) void helix_main(const float* __restrict__ x,
                                                       const float* __restrict__ b_enc,
                                                       const float* __restrict__ omegas,
                                                       const float* __restrict__ K_globals,
                                                       const float* __restrict__ b_dec,
                                                       const _Float16* __restrict__ ws,
                                                       float* __restrict__ out) {
  const int tid   = threadIdx.x;
  const int lane  = tid & 31;
  const int wave  = tid >> 5;       // 0..7
  const int l16   = lane & 15;
  const int khalf = lane >> 4;      // 0/1
  const int ntb   = wave * 2;       // base N tile (2 per wave)
  const int row0  = blockIdx.x * BM;

  __shared__ _Float16 Sh[BM * LDSTRIDE];   // sin(theta) f16           ( 8448 B)
  __shared__ _Float16 Ch[BM * LDSTRIDE];   // cos(theta) f16           ( 8448 B)
  __shared__ _Float16 KBl[KB_LAYER];       // one layer's packed K_eff (131072 B)
  __shared__ float sumc[BM], sums[BM];

  const _Float16* encB = ws + ENCB_OFF;
  const _Float16* decB = ws + DECB_OFF;

  float* out_y     = out;                          // [2048,128]
  float* out_theta = out + 2048 * OUT_DIM;         // [2048,256]
  float* out_coh   = out_theta + 2048 * NOSC;      // [2048]

  // ---------------- encoder: theta0 = wrap(x @ W_enc^T + b_enc), WMMA over 4 K-chunks
  v8f acc[2];
  acc[0] = (v8f){0,0,0,0,0,0,0,0};
  acc[1] = (v8f){0,0,0,0,0,0,0,0};

  for (int c = 0; c < 4; ++c) {
    __syncthreads();
    for (int i = tid; i < BM * 128; i += NTHREADS) {  // x chunk -> Sh (f16)
      int m = i >> 7, k = i & 127;
      Sh[m * LDSTRIDE + k] = (_Float16)x[(size_t)(row0 + m) * IN_DIM + c * 128 + k];
    }
    __syncthreads();
    #pragma unroll
    for (int kt = 0; kt < 4; ++kt) {
      v16h a = load_A(Sh, l16, kt * 32, khalf);
      #pragma unroll
      for (int t = 0; t < 2; ++t) {
        v16h b = load_B(encB, (c * 4 + kt) * 16 + (ntb + t), lane);
        acc[t] = wmma_f16(a, b, acc[t]);
      }
    }
  }

  float th[2][8];
  #pragma unroll
  for (int t = 0; t < 2; ++t) {
    int n = (ntb + t) * 16 + l16;
    float be = b_enc[n];
    #pragma unroll
    for (int r = 0; r < 8; ++r) th[t][r] = wrapf(acc[t][r] + be);
  }

  // ---------------- Kuramoto layers ----------------
  for (int L = 0; L < NLAYERS; ++L) {
    // stage this layer's packed K_eff into LDS via async DMA (reused for 10 steps)
    __syncthreads();   // previous layer's GEMM reads of KBl complete
    {
      const char* src = (const char*)(ws + KB_OFF + L * KB_LAYER);
      for (int i = tid; i < (KB_LAYER * 2) / 16; i += NTHREADS) {
        unsigned ldsoff = (unsigned)(uintptr_t)(&KBl[0]) + (unsigned)(i * 16);
        unsigned long long ga = (unsigned long long)(uintptr_t)(src + (size_t)i * 16);
        asm volatile("global_load_async_to_lds_b128 %0, %1, off"
                     :: "v"(ldsoff), "v"(ga) : "memory");
      }
      asm volatile("s_wait_asynccnt 0x0" ::: "memory");
    }
    float om[2];
    #pragma unroll
    for (int t = 0; t < 2; ++t) om[t] = omegas[L * NOSC + (ntb + t) * 16 + l16];
    const float g = K_globals[L] * (1.0f / (float)NOSC);

    for (int s = 0; s < NSTEPS; ++s) {
      __syncthreads();   // KBl staged / previous step's LDS reads complete
      // phase 1: sincos -> registers + f16 copies into LDS (A-matrix staging)
      float sreg[2][8], creg[2][8];
      #pragma unroll
      for (int t = 0; t < 2; ++t) {
        int n = (ntb + t) * 16 + l16;
        #pragma unroll
        for (int r = 0; r < 8; ++r) {
          int m = r + khalf * 8;
          float si, co;
          __sincosf(th[t][r], &si, &co);
          sreg[t][r] = si; creg[t][r] = co;
          Sh[m * LDSTRIDE + n] = (_Float16)si;
          Ch[m * LDSTRIDE + n] = (_Float16)co;
        }
      }
      __syncthreads();
      // phase 2: SK = S@K_eff, CK = C@K_eff  (A and B both from LDS)
      v8f as[2], ac[2];
      as[0] = (v8f){0,0,0,0,0,0,0,0}; as[1] = (v8f){0,0,0,0,0,0,0,0};
      ac[0] = (v8f){0,0,0,0,0,0,0,0}; ac[1] = (v8f){0,0,0,0,0,0,0,0};
      #pragma unroll
      for (int kt = 0; kt < 8; ++kt) {
        v16h aS = load_A(Sh, l16, kt * 32, khalf);
        v16h aC = load_A(Ch, l16, kt * 32, khalf);
        #pragma unroll
        for (int t = 0; t < 2; ++t) {
          v16h b = load_B(KBl, kt * 16 + (ntb + t), lane);
          as[t] = wmma_f16(aS, b, as[t]);
          ac[t] = wmma_f16(aC, b, ac[t]);
        }
      }
      // phase 3: theta update (coupling = c*SK - s*CK) from registers, wrap
      #pragma unroll
      for (int t = 0; t < 2; ++t) {
        #pragma unroll
        for (int r = 0; r < 8; ++r) {
          float coup = creg[t][r] * as[t][r] - sreg[t][r] * ac[t][r];
          th[t][r] = wrapf(th[t][r] + DTC * (om[t] + g * coup));
        }
      }
    }
  }

  // ---------------- theta output ----------------
  #pragma unroll
  for (int t = 0; t < 2; ++t) {
    int n = (ntb + t) * 16 + l16;
    #pragma unroll
    for (int r = 0; r < 8; ++r) {
      int m = r + khalf * 8;
      out_theta[(size_t)(row0 + m) * NOSC + n] = th[t][r];
    }
  }

  // ---------------- coherence: sqrt(mean_cos^2 + mean_sin^2) per row ----------------
  __syncthreads();                       // prior LDS reads done
  if (tid < BM) { sumc[tid] = 0.0f; sums[tid] = 0.0f; }
  __syncthreads();
  #pragma unroll
  for (int t = 0; t < 2; ++t) {
    #pragma unroll
    for (int r = 0; r < 8; ++r) {
      int m = r + khalf * 8;
      float si, co;
      __sincosf(th[t][r], &si, &co);
      atomicAdd(&sums[m], si);
      atomicAdd(&sumc[m], co);
    }
  }
  __syncthreads();
  if (tid < BM) {
    float mc = sumc[tid] * (1.0f / (float)NOSC);
    float ms = sums[tid] * (1.0f / (float)NOSC);
    out_coh[row0 + tid] = sqrtf(mc * mc + ms * ms);
  }

  // ---------------- decoder: out = theta @ W_dec^T + b_dec (WMMA) ----------------
  // stage final theta as f16 A-matrix into Sh
  #pragma unroll
  for (int t = 0; t < 2; ++t) {
    int n = (ntb + t) * 16 + l16;
    #pragma unroll
    for (int r = 0; r < 8; ++r) {
      int m = r + khalf * 8;
      Sh[m * LDSTRIDE + n] = (_Float16)th[t][r];
    }
  }
  __syncthreads();
  // 8 output N-tiles over 8 waves -> 1 tile per wave
  v8f oacc = (v8f){0,0,0,0,0,0,0,0};
  #pragma unroll
  for (int kt = 0; kt < 8; ++kt) {
    v16h a = load_A(Sh, l16, kt * 32, khalf);
    v16h b = load_B(decB, kt * 8 + wave, lane);
    oacc = wmma_f16(a, b, oacc);
  }
  {
    int o = wave * 16 + l16;
    float bd = b_dec[o];
    #pragma unroll
    for (int r = 0; r < 8; ++r) {
      int m = r + khalf * 8;
      out_y[(size_t)(row0 + m) * OUT_DIM + o] = oacc[r] + bd;
    }
  }
}

extern "C" void kernel_launch(void* const* d_in, const int* in_sizes, int n_in,
                              void* d_out, int out_size, void* d_ws, size_t ws_size,
                              hipStream_t stream) {
  (void)in_sizes; (void)n_in; (void)out_size; (void)ws_size;
  const float* x         = (const float*)d_in[0];
  const float* W_enc     = (const float*)d_in[1];
  const float* b_enc     = (const float*)d_in[2];
  const float* Ks        = (const float*)d_in[3];
  const float* omegas    = (const float*)d_in[4];
  const float* K_globals = (const float*)d_in[5];
  const float* mu_gates  = (const float*)d_in[6];
  const float* W_dec     = (const float*)d_in[7];
  const float* b_dec     = (const float*)d_in[8];
  _Float16* ws  = (_Float16*)d_ws;      // needs 832 KB scratch
  float*    out = (float*)d_out;

  // pack W_enc^T / K_eff / W_dec^T into WMMA B-register layout (f16) once
  helix_prep<<<(WS_HALVES + 255) / 256, 256, 0, stream>>>(W_enc, Ks, mu_gates, W_dec, ws);
  // 2048 rows / 16 per block = 128 blocks x 256 threads (8 waves)
  helix_main<<<2048 / BM, NTHREADS, 0, stream>>>(x, b_enc, omegas, K_globals, b_dec, ws, out);
}
